// _MMD_loss__B_38646115730033
// MI455X (gfx1250) — matrix operand order repaired and checked by reading, not compiled, not run on hardware
//
#include <hip/hip_runtime.h>
#include <hip/hip_bf16.h>

typedef __attribute__((ext_vector_type(8)))  float  v8f;
typedef __attribute__((ext_vector_type(8)))  __bf16 v8bf;
typedef __attribute__((ext_vector_type(16))) __bf16 v16bf;

#define D_DIM 256
#define BT 128          // block tile (M and N)
#define KC 32           // K chunk staged in LDS == WMMA K
#define LSB 40          // bf16 row stride: 32 + 8 pad -> 80B rows, conflict-free b128 frags

// ---------------- prep: weights + row squared-norms ----------------
__global__ __launch_bounds__(256) void mmd_prep_kernel(
    const float* __restrict__ S, const int* __restrict__ label,
    const float* __restrict__ T, const float* __restrict__ cw,
    float* __restrict__ w, float* __restrict__ rns, float* __restrict__ rnt,
    int NS, int NT)
{
    int i = blockIdx.x * blockDim.x + threadIdx.x;
    if (i < NS) {
        w[i] = cw[label[i]];
        const float4* row = (const float4*)(S + (size_t)i * D_DIM);
        float acc = 0.f;
        #pragma unroll 8
        for (int k = 0; k < D_DIM / 4; ++k) {
            float4 v = row[k];
            acc += v.x * v.x + v.y * v.y + v.z * v.z + v.w * v.w;
        }
        rns[i] = acc;
    }
    if (i < NT) {
        const float4* row = (const float4*)(T + (size_t)i * D_DIM);
        float acc = 0.f;
        #pragma unroll 8
        for (int k = 0; k < D_DIM / 4; ++k) {
            float4 v = row[k];
            acc += v.x * v.x + v.y * v.y + v.z * v.z + v.w * v.w;
        }
        rnt[i] = acc;
    }
}

// ---------------- deterministic sum of w -> n ----------------
__global__ __launch_bounds__(256) void mmd_sumw_kernel(
    const float* __restrict__ w, float* __restrict__ nout, int NS)
{
    __shared__ float sm[256];
    float s = 0.f;
    for (int i = threadIdx.x; i < NS; i += 256) s += w[i];
    sm[threadIdx.x] = s;
    __syncthreads();
    for (int off = 128; off > 0; off >>= 1) {
        if ((int)threadIdx.x < off) sm[threadIdx.x] += sm[threadIdx.x + off];
        __syncthreads();
    }
    if (threadIdx.x == 0) nout[0] = sm[0];
}

// split 8 floats -> bf16 hi + bf16 lo (lo = RN_bf16(x - (float)hi), x-hi exact in fp32)
__device__ __forceinline__ void split8(const float* f, v8bf& h, v8bf& l) {
    #pragma unroll
    for (int q = 0; q < 8; ++q) {
        float x = f[q];
        __bf16 hb = (__bf16)x;
        h[q] = hb;
        l[q] = (__bf16)(x - (float)hb);
    }
}

// assemble A fragment: two contiguous 8-elem chunks at +0 and +16 bf16
__device__ __forceinline__ v16bf afrag(const __bf16* p) {
    v8bf c0 = *(const v8bf*)p;
    v8bf c1 = *(const v8bf*)(p + 16);
    return __builtin_shufflevector(c0, c1, 0,1,2,3,4,5,6,7,8,9,10,11,12,13,14,15);
}

// ---------------- main: tiled 3xBF16 WMMA Gram + fused exp/weight reduce ----------------
__global__ __launch_bounds__(256) void mmd_tile_kernel(
    const float* __restrict__ S, const float* __restrict__ T,
    const float* __restrict__ w, const float* __restrict__ rns,
    const float* __restrict__ rnt, const float* __restrict__ sigma,
    float* __restrict__ partials)
{
    __shared__ __bf16 Ah[BT * LSB];
    __shared__ __bf16 Al[BT * LSB];
    __shared__ __bf16 Bh[BT * LSB];
    __shared__ __bf16 Bl[BT * LSB];
    __shared__ float  wsum[8];

    const int z = blockIdx.z;           // 0: S.S (w,w)  1: T.T (1,1)  2: S.T (w,1)
    const float* X   = (z == 1) ? T : S;
    const float* Y   = (z == 0) ? S : T;
    const float* rnA = (z == 1) ? rnt : rns;
    const float* rnB = (z == 0) ? rns : rnt;
    const int useWA = (z != 1);
    const int useWB = (z == 0);

    const int rowBase = blockIdx.y * BT;
    const int colBase = blockIdx.x * BT;

    const int tid  = threadIdx.x;
    const int wid  = tid >> 5;          // wave 0..7
    const int lane = tid & 31;
    const int half = lane >> 4;         // 0/1
    const int lm   = lane & 15;
    const int wrow = wid & 3;           // 4 waves along M: 32 rows each
    const int wcol = wid >> 2;          // 2 waves along N: 64 cols each

    v8f acc[2][4] = {};

    // cooperative global->LDS: each thread owns 16 contiguous floats of one row per panel
    const int lr = tid >> 1;            // 0..127
    const int lc = (tid & 1) * 16;      // 0 or 16
    const float* gA = X + (size_t)(rowBase + lr) * D_DIM + lc;
    const float* gB = Y + (size_t)(colBase + lr) * D_DIM + lc;
    const int lofs = lr * LSB + lc;

    // fragment base offsets (bf16 elements)
    const int aoff0 = (wrow * 32 +  0 + lm) * LSB + 8 * half;   // A: K chunks at +0,+16
    const int aoff1 = (wrow * 32 + 16 + lm) * LSB + 8 * half;
    const int boffb = (wcol * 64 + lm) * LSB + 8 * half;        // B: K chunks at +0,+16

    for (int kc = 0; kc < D_DIM; kc += KC) {
        __syncthreads();
        {
            float fa[16], fb[16];
            #pragma unroll
            for (int q = 0; q < 16; q += 4) {
                *(float4*)&fa[q] = *(const float4*)(gA + kc + q);
                *(float4*)&fb[q] = *(const float4*)(gB + kc + q);
            }
            v8bf h0, h1, l0, l1;
            split8(&fa[0], h0, l0); split8(&fa[8], h1, l1);
            *(v8bf*)&Ah[lofs]     = h0; *(v8bf*)&Ah[lofs + 8] = h1;
            *(v8bf*)&Al[lofs]     = l0; *(v8bf*)&Al[lofs + 8] = l1;
            split8(&fb[0], h0, l0); split8(&fb[8], h1, l1);
            *(v8bf*)&Bh[lofs]     = h0; *(v8bf*)&Bh[lofs + 8] = h1;
            *(v8bf*)&Bl[lofs]     = l0; *(v8bf*)&Bl[lofs + 8] = l1;
        }
        __syncthreads();

        // A fragments (16x32 bf16 layout): lane holds K in {8h..8h+7} u {16+8h..16+8h+7}
        v16bf a0h = afrag(&Ah[aoff0]);
        v16bf a0l = afrag(&Al[aoff0]);
        v16bf a1h = afrag(&Ah[aoff1]);
        v16bf a1l = afrag(&Al[aoff1]);

        #pragma unroll
        for (int j = 0; j < 4; ++j) {
            // B fragment (32x16 bf16): lane holds Y[n][K in {8h..8h+7} u {16+8h..16+8h+7}]
            const int bo = boffb + j * 16 * LSB;
            v16bf bh = afrag(&Bh[bo]);
            v16bf bl = afrag(&Bl[bo]);

            acc[0][j] = __builtin_amdgcn_wmma_f32_16x16x32_bf16(false, a0h, false, bh, (short)0, acc[0][j], false, false);
            acc[0][j] = __builtin_amdgcn_wmma_f32_16x16x32_bf16(false, a0h, false, bl, (short)0, acc[0][j], false, false);
            acc[0][j] = __builtin_amdgcn_wmma_f32_16x16x32_bf16(false, a0l, false, bh, (short)0, acc[0][j], false, false);
            acc[1][j] = __builtin_amdgcn_wmma_f32_16x16x32_bf16(false, a1h, false, bh, (short)0, acc[1][j], false, false);
            acc[1][j] = __builtin_amdgcn_wmma_f32_16x16x32_bf16(false, a1h, false, bl, (short)0, acc[1][j], false, false);
            acc[1][j] = __builtin_amdgcn_wmma_f32_16x16x32_bf16(false, a1l, false, bh, (short)0, acc[1][j], false, false);
        }
    }

    // fused epilogue: K = exp(-(ra + rb - 2*dot) * inv), weighted sum
    const float inv = 1.0f / (2.0f * sigma[0]);
    float lsum = 0.f;
    #pragma unroll
    for (int i = 0; i < 2; ++i) {
        const int rg0 = rowBase + wrow * 32 + i * 16 + 8 * half;  // C layout: m = v + 8*half
        float ra[8], wa[8];
        #pragma unroll
        for (int v = 0; v < 8; ++v) {
            ra[v] = rnA[rg0 + v];
            wa[v] = useWA ? w[rg0 + v] : 1.0f;
        }
        #pragma unroll
        for (int j = 0; j < 4; ++j) {
            const int cg = colBase + wcol * 64 + j * 16 + lm;     // C layout: n = lane%16
            const float rb = rnB[cg];
            const float wb = useWB ? w[cg] : 1.0f;
            #pragma unroll
            for (int v = 0; v < 8; ++v) {
                float kv = __expf(-(ra[v] + rb - 2.0f * acc[i][j][v]) * inv);
                lsum += wa[v] * wb * kv;
            }
        }
    }

    // wave32 reduce, then cross-wave via LDS; one unique partial per block (no atomics)
    #pragma unroll
    for (int off = 16; off > 0; off >>= 1) lsum += __shfl_xor(lsum, off, 32);
    if (lane == 0) wsum[wid] = lsum;
    __syncthreads();
    if (tid == 0) {
        float s = 0.f;
        #pragma unroll
        for (int q = 0; q < 8; ++q) s += wsum[q];
        partials[(z * gridDim.y + blockIdx.y) * gridDim.x + blockIdx.x] = s;
    }
}

// ---------------- final: deterministic reduce + scalar combine ----------------
__global__ __launch_bounds__(256) void mmd_final_kernel(
    const float* __restrict__ partials, const float* __restrict__ nptr,
    float* __restrict__ out, int nblocks, int NT)
{
    __shared__ float sm[256];
    float S[3];
    for (int zz = 0; zz < 3; ++zz) {
        float s = 0.f;
        for (int i = threadIdx.x; i < nblocks; i += 256) s += partials[zz * nblocks + i];
        sm[threadIdx.x] = s;
        __syncthreads();
        for (int off = 128; off > 0; off >>= 1) {
            if ((int)threadIdx.x < off) sm[threadIdx.x] += sm[threadIdx.x + off];
            __syncthreads();
        }
        S[zz] = sm[0];
        __syncthreads();
    }
    if (threadIdx.x == 0) {
        float n  = nptr[0];
        float nt = (float)NT;
        out[0] = S[0] / (n * n) + S[1] / (nt * nt) - 2.0f * S[2] / (n * nt);
    }
}

extern "C" void kernel_launch(void* const* d_in, const int* in_sizes, int n_in,
                              void* d_out, int out_size, void* d_ws, size_t ws_size,
                              hipStream_t stream) {
    const float* S     = (const float*)d_in[0];
    const int*   label = (const int*)d_in[1];
    const float* T     = (const float*)d_in[2];
    const float* cw    = (const float*)d_in[3];
    const float* sigma = (const float*)d_in[4];

    const int NS = in_sizes[1];
    const int NT = in_sizes[2] / D_DIM;

    float* ws       = (float*)d_ws;
    float* w        = ws;                 // [NS]
    float* rns      = w + NS;             // [NS]
    float* rnt      = rns + NS;           // [NT]
    float* nval     = rnt + NT;           // [1] (+pad)
    float* partials = nval + 64;          // [3 * (NS/BT)^2]

    const int nmax = (NS > NT) ? NS : NT;
    mmd_prep_kernel<<<(nmax + 255) / 256, 256, 0, stream>>>(S, label, T, cw, w, rns, rnt, NS, NT);
    mmd_sumw_kernel<<<1, 256, 0, stream>>>(w, nval, NS);

    const int nb = NS / BT;               // NS == NT == 4096 -> 32
    dim3 grid(nb, nb, 3);
    mmd_tile_kernel<<<grid, 256, 0, stream>>>(S, T, w, rns, rnt, sigma, partials);

    mmd_final_kernel<<<1, 256, 0, stream>>>(partials, nval, (float*)d_out, nb * nb, NT);
}
